// ResGIN_22247930594064
// MI455X (gfx1250) — compile-verified, matrix operand order checked by reading, not run on hardware
//
#include <hip/hip_runtime.h>
#include <hip/hip_bf16.h>
#include <math.h>

typedef __attribute__((ext_vector_type(16))) __bf16 v16bf;
typedef __attribute__((ext_vector_type(8)))  __bf16 v8bf;
typedef __attribute__((ext_vector_type(4)))  __bf16 v4bf;
typedef __attribute__((ext_vector_type(8)))  float  v8f;

#define STR 40        // LDS row stride in bf16 elems (multiple of 8 -> 16B aligned rows)
#define DMODEL 128

// ---- WMMA fragment loader (A as MxK rows, B as NxK "columns", both K-contiguous in LDS)
// Layout per CDNA5 ISA 16-bit A 16x32:
//   lanes 0-15  : row = lane,    elems 0..7 = K 0..7,  elems 8..15 = K 16..23
//   lanes 16-31 : row = lane-16, elems 0..7 = K 8..15, elems 8..15 = K 24..31
__device__ __forceinline__ v16bf frag_ld(const __bf16* tile, int rowBase, int lane) {
    int r  = rowBase + (lane & 15);
    int hi = (lane >> 4) & 1;
    const __bf16* p = tile + r * STR + hi * 8;
    v8bf lo = *(const v8bf*)(p);
    v8bf hh = *(const v8bf*)(p + 16);
    return __builtin_shufflevector(lo, hh, 0,1,2,3,4,5,6,7,8,9,10,11,12,13,14,15);
}

// ---- async global -> LDS copy of one 16B chunk (CDNA5 TDM-lite path, ASYNCcnt-tracked)
__device__ __forceinline__ void async_ld_b128(unsigned lds_byte_addr, const void* gptr) {
    unsigned long long ga = (unsigned long long)gptr;
    asm volatile("global_load_async_to_lds_b128 %0, %1, off"
                 :: "v"(lds_byte_addr), "v"(ga) : "memory");
}
__device__ __forceinline__ void async_wait0() {
    asm volatile("s_wait_asynccnt 0x0" ::: "memory");
}

__global__ void k_zero4(float4* p, long long n4) {
    long long i = (long long)blockIdx.x * blockDim.x + threadIdx.x;
    long long s = (long long)gridDim.x * blockDim.x;
    float4 z = {0.f, 0.f, 0.f, 0.f};
    for (; i < n4; i += s) p[i] = z;
}

// convert f32 [K][Nc] -> bf16 transposed [Nc][K]
__global__ void k_cvt_t(const float* __restrict__ s, __bf16* __restrict__ d, int K, int Nc) {
    int i = blockIdx.x * blockDim.x + threadIdx.x;
    if (i < K * Nc) {
        int k = i / Nc, c = i - k * Nc;
        d[(size_t)c * K + k] = (__bf16)s[i];
    }
}

// ---- h = x @ embW + emb_b   [N,FIN]x[FIN,128]; wT = bf16 [128][FIN]
__global__ __launch_bounds__(256)
void k_emb(const float* __restrict__ x, const __bf16* __restrict__ wT,
           const float* __restrict__ bias, float* __restrict__ h, int Nn, int K) {
    __shared__ __align__(16) __bf16 As[128 * STR];
    __shared__ __align__(16) __bf16 Bs[128 * STR];
    int tid = threadIdx.x, lane = tid & 31, wv = tid >> 5;
    int row0 = blockIdx.x * 128;
    unsigned bs_base = (unsigned)(unsigned long long)(const void*)&Bs[0];
    v8f vzero = {0.f,0.f,0.f,0.f,0.f,0.f,0.f,0.f};
    v8f acc[8];
#pragma unroll
    for (int i = 0; i < 8; i++) acc[i] = vzero;
    for (int kt = 0; kt < K; kt += 32) {
        // B: contiguous 16B async copies from transposed weights
        for (int t = tid; t < 128 * 4; t += 256) {
            int c = t >> 2, q = t & 3;
            async_ld_b128(bs_base + (unsigned)((c * STR + q * 8) * 2),
                          wT + (size_t)c * K + kt + q * 8);
        }
        // A: f32 -> bf16, 4 elems/thread
        for (int t = tid; t < 128 * 8; t += 256) {
            int r = t >> 3, q = t & 7, gr = row0 + r;
            float4 xv = {0.f, 0.f, 0.f, 0.f};
            if (gr < Nn) xv = *(const float4*)(x + (size_t)gr * K + kt + q * 4);
            v4bf p = {(__bf16)xv.x, (__bf16)xv.y, (__bf16)xv.z, (__bf16)xv.w};
            *(v4bf*)(&As[r * STR + q * 4]) = p;
        }
        async_wait0();
        __syncthreads();
        v16bf a = frag_ld(As, wv * 16, lane);
#pragma unroll
        for (int ct = 0; ct < 8; ct++) {
            v16bf b = frag_ld(Bs, ct * 16, lane);
            acc[ct] = __builtin_amdgcn_wmma_f32_16x16x32_bf16(false, a, false, b,
                                                              (short)0, acc[ct], false, false);
        }
        __syncthreads();
    }
    int n = lane & 15, hi = lane >> 4;
#pragma unroll
    for (int ct = 0; ct < 8; ct++) {
        int col = ct * 16 + n;
        float bb = bias[col];
#pragma unroll
        for (int v = 0; v < 8; v++) {
            int row = row0 + wv * 16 + v + 8 * hi;
            if (row < Nn) h[(size_t)row * 128 + col] = acc[ct][v] + bb;
        }
    }
}

// ---- agg[dst] += h[src] over edges: one wave per edge, 4 floats/lane, HW f32 atomics
__global__ __launch_bounds__(256)
void k_edge(const int* __restrict__ ei, const float* __restrict__ h,
            float* __restrict__ agg, int E_) {
    int lane = threadIdx.x & 31;
    int wid = (blockIdx.x * blockDim.x + threadIdx.x) >> 5;
    int nw  = (gridDim.x * blockDim.x) >> 5;
    for (int e = wid; e < E_; e += nw) {
        int s = ei[e], d = ei[E_ + e];
        const float4 v = *(const float4*)(h + (size_t)s * DMODEL + lane * 4);
        float* ap = agg + (size_t)d * DMODEL + lane * 4;
        unsafeAtomicAdd(ap + 0, v.x);
        unsafeAtomicAdd(ap + 1, v.y);
        unsafeAtomicAdd(ap + 2, v.z);
        unsafeAtomicAdd(ap + 3, v.w);
    }
}

// ---- u = ((1+eps)h + agg) @ W1 + b1 ; per-col sum/sumsq for BN. wT1 = bf16 [256][128]
__global__ __launch_bounds__(256)
void k_mlp1(const float* __restrict__ h, const float* __restrict__ agg,
            const __bf16* __restrict__ wT1, const float* __restrict__ b1,
            const float* __restrict__ epsp, int layer,
            float* __restrict__ u, float* __restrict__ stats, int Nn) {
    __shared__ __align__(16) __bf16 As[128 * STR];
    __shared__ __align__(16) __bf16 Bs[256 * STR];
    int tid = threadIdx.x, lane = tid & 31, wv = tid >> 5;
    int row0 = blockIdx.x * 128;
    unsigned bs_base = (unsigned)(unsigned long long)(const void*)&Bs[0];
    float ep1 = 1.f + epsp[layer];
    v8f vzero = {0.f,0.f,0.f,0.f,0.f,0.f,0.f,0.f};
    v8f acc[16];
#pragma unroll
    for (int i = 0; i < 16; i++) acc[i] = vzero;
    for (int kt = 0; kt < 128; kt += 32) {
        for (int t = tid; t < 256 * 4; t += 256) {
            int c = t >> 2, q = t & 3;
            async_ld_b128(bs_base + (unsigned)((c * STR + q * 8) * 2),
                          wT1 + (size_t)c * 128 + kt + q * 8);
        }
        for (int t = tid; t < 128 * 8; t += 256) {
            int r = t >> 3, q = t & 7, gr = row0 + r;
            float4 hv = {0.f,0.f,0.f,0.f}, av = {0.f,0.f,0.f,0.f};
            if (gr < Nn) {
                size_t idx = (size_t)gr * 128 + kt + q * 4;
                hv = *(const float4*)(h + idx);
                av = *(const float4*)(agg + idx);
            }
            v4bf p = {(__bf16)fmaf(ep1, hv.x, av.x), (__bf16)fmaf(ep1, hv.y, av.y),
                      (__bf16)fmaf(ep1, hv.z, av.z), (__bf16)fmaf(ep1, hv.w, av.w)};
            *(v4bf*)(&As[r * STR + q * 4]) = p;
        }
        async_wait0();
        __syncthreads();
        v16bf a = frag_ld(As, wv * 16, lane);
#pragma unroll
        for (int ct = 0; ct < 16; ct++) {
            v16bf b = frag_ld(Bs, ct * 16, lane);
            acc[ct] = __builtin_amdgcn_wmma_f32_16x16x32_bf16(false, a, false, b,
                                                              (short)0, acc[ct], false, false);
        }
        __syncthreads();
    }
    int n = lane & 15, hi = lane >> 4;
#pragma unroll
    for (int ct = 0; ct < 16; ct++) {
        int col = ct * 16 + n;
        float bb = b1[col];
        float s = 0.f, sq = 0.f;
#pragma unroll
        for (int v = 0; v < 8; v++) {
            int row = row0 + wv * 16 + v + 8 * hi;
            if (row < Nn) {
                float val = acc[ct][v] + bb;
                u[(size_t)row * 256 + col] = val;
                s += val; sq += val * val;
            }
        }
        s  += __shfl_xor(s, 16, 32);
        sq += __shfl_xor(sq, 16, 32);
        if (lane < 16) {
            unsafeAtomicAdd(&stats[col], s);
            unsafeAtomicAdd(&stats[256 + col], sq);
        }
    }
}

__global__ void k_bnfin(const float* __restrict__ stats, const float* __restrict__ g,
                        const float* __restrict__ be, float* __restrict__ ss, float invN) {
    int i = threadIdx.x;           // 256 threads
    float mu  = stats[i] * invN;
    float var = stats[256 + i] * invN - mu * mu;
    float sc  = g[i] * rsqrtf(var + 1e-5f);
    ss[i]       = sc;
    ss[256 + i] = be[i] - mu * sc;
}

// ---- z2 = relu(BN(u)) @ W2 + b2 ; h = relu(h + z2) in place. wT2 = bf16 [128][256]
__global__ __launch_bounds__(256)
void k_mlp2(const float* __restrict__ u, const float* __restrict__ ss,
            const __bf16* __restrict__ wT2, const float* __restrict__ b2,
            float* __restrict__ h, int Nn) {
    __shared__ __align__(16) __bf16 As[128 * STR];
    __shared__ __align__(16) __bf16 Bs[128 * STR];
    int tid = threadIdx.x, lane = tid & 31, wv = tid >> 5;
    int row0 = blockIdx.x * 128;
    unsigned bs_base = (unsigned)(unsigned long long)(const void*)&Bs[0];
    v8f vzero = {0.f,0.f,0.f,0.f,0.f,0.f,0.f,0.f};
    v8f acc[8];
#pragma unroll
    for (int i = 0; i < 8; i++) acc[i] = vzero;
    for (int kt = 0; kt < 256; kt += 32) {
        for (int t = tid; t < 128 * 4; t += 256) {
            int c = t >> 2, q = t & 3;
            async_ld_b128(bs_base + (unsigned)((c * STR + q * 8) * 2),
                          wT2 + (size_t)c * 256 + kt + q * 8);
        }
        for (int t = tid; t < 128 * 8; t += 256) {
            int r = t >> 3, q = t & 7, gr = row0 + r;
            int k0 = kt + q * 4;
            float4 uv = {0.f,0.f,0.f,0.f};
            if (gr < Nn) uv = *(const float4*)(u + (size_t)gr * 256 + k0);
            float4 sc = *(const float4*)(ss + k0);
            float4 sh = *(const float4*)(ss + 256 + k0);
            v4bf p = {(__bf16)fmaxf(fmaf(uv.x, sc.x, sh.x), 0.f),
                      (__bf16)fmaxf(fmaf(uv.y, sc.y, sh.y), 0.f),
                      (__bf16)fmaxf(fmaf(uv.z, sc.z, sh.z), 0.f),
                      (__bf16)fmaxf(fmaf(uv.w, sc.w, sh.w), 0.f)};
            *(v4bf*)(&As[r * STR + q * 4]) = p;
        }
        async_wait0();
        __syncthreads();
        v16bf a = frag_ld(As, wv * 16, lane);
#pragma unroll
        for (int ct = 0; ct < 8; ct++) {
            v16bf b = frag_ld(Bs, ct * 16, lane);
            acc[ct] = __builtin_amdgcn_wmma_f32_16x16x32_bf16(false, a, false, b,
                                                              (short)0, acc[ct], false, false);
        }
        __syncthreads();
    }
    int n = lane & 15, hi = lane >> 4;
#pragma unroll
    for (int ct = 0; ct < 8; ct++) {
        int col = ct * 16 + n;
        float bb = b2[col];
#pragma unroll
        for (int v = 0; v < 8; v++) {
            int row = row0 + wv * 16 + v + 8 * hi;
            if (row < Nn) {
                size_t idx = (size_t)row * 128 + col;
                h[idx] = fmaxf(h[idx] + acc[ct][v] + bb, 0.f);
            }
        }
    }
}

// ---- segment sum + count (mean pool), one wave per node
__global__ __launch_bounds__(256)
void k_pool(const float* __restrict__ h, const int* __restrict__ batch,
            float* __restrict__ pooled, float* __restrict__ cnt, int Nn) {
    int lane = threadIdx.x & 31;
    int wid = (blockIdx.x * blockDim.x + threadIdx.x) >> 5;
    int nw  = (gridDim.x * blockDim.x) >> 5;
    for (int nI = wid; nI < Nn; nI += nw) {
        int g = batch[nI];
        const float4 v = *(const float4*)(h + (size_t)nI * DMODEL + lane * 4);
        float* pp = pooled + (size_t)g * DMODEL + lane * 4;
        unsafeAtomicAdd(pp + 0, v.x);
        unsafeAtomicAdd(pp + 1, v.y);
        unsafeAtomicAdd(pp + 2, v.z);
        unsafeAtomicAdd(pp + 3, v.w);
        if (lane == 0) unsafeAtomicAdd(cnt + g, 1.f);
    }
}

__global__ void k_poolfin(const float* __restrict__ pooled, const float* __restrict__ cnt,
                          float* __restrict__ xc, int G_, int off) {
    int i = blockIdx.x * blockDim.x + threadIdx.x;
    if (i < G_ * 128) {
        int g = i >> 7, d = i & 127;
        xc[(size_t)g * 256 + off + d] = pooled[i] / fmaxf(cnt[g], 1.f);
    }
}

// ---- tiny MLP head: fc1(256)->relu->fc2(64)->relu->out(1)->sigmoid, one block
__global__ __launch_bounds__(256)
void k_head(const float* __restrict__ xc,
            const float* __restrict__ fc1W, const float* __restrict__ fc1b,
            const float* __restrict__ fc2W, const float* __restrict__ fc2b,
            const float* __restrict__ oW, const float* __restrict__ ob,
            float* __restrict__ t1, float* __restrict__ t2,
            float* __restrict__ out, int G_) {
    int tid = threadIdx.x;
    {   // stage 1: column tid of fc1 for all graphs
        int j = tid;
        for (int g = 0; g < G_; g++) {
            float s = fc1b[j];
            for (int k = 0; k < 256; k++) s += xc[(size_t)g * 256 + k] * fc1W[(size_t)k * 256 + j];
            t1[(size_t)g * 256 + j] = fmaxf(s, 0.f);
        }
    }
    __syncthreads();
    {   // stage 2
        int j = tid & 63, g0 = (tid >> 6) * 16;
        for (int g = g0; g < g0 + 16; g++) {
            if (g < G_) {
                float s = fc2b[j];
                for (int k = 0; k < 256; k++) s += t1[(size_t)g * 256 + k] * fc2W[(size_t)k * 64 + j];
                t2[(size_t)g * 64 + j] = fmaxf(s, 0.f);
            }
        }
    }
    __syncthreads();
    if (tid < G_) {
        float s = ob[0];
        for (int k = 0; k < 64; k++) s += t2[(size_t)tid * 64 + k] * oW[k];
        out[tid] = 1.f / (1.f + expf(-s));
    }
}

extern "C" void kernel_launch(void* const* d_in, const int* in_sizes, int n_in,
                              void* d_out, int out_size, void* d_ws, size_t ws_size,
                              hipStream_t stream) {
    (void)n_in; (void)ws_size;
    const int Nn  = in_sizes[2];
    const int E_  = in_sizes[1] / 2;
    const int FIN = in_sizes[0] / Nn;
    const int G_  = out_size;
    const int L_  = 4;

    char* ws = (char*)d_ws;
    size_t off = 0;
    auto take = [&](size_t bytes) { size_t o = (off + 255) & ~(size_t)255; off = o + bytes; return o; };

    size_t o_wbemb[2], o_wbw1[2], o_wbw2[2];
    for (int b = 0; b < 2; b++) {
        o_wbemb[b] = take((size_t)FIN * 128 * 2);     // transposed [128][FIN]
        o_wbw1[b]  = take((size_t)L_ * 128 * 256 * 2); // per layer [256][128]
        o_wbw2[b]  = take((size_t)L_ * 256 * 128 * 2); // per layer [128][256]
    }
    size_t o_h   = take((size_t)Nn * 128 * 4);
    size_t o_agg = take((size_t)Nn * 128 * 4);
    size_t o_u   = take((size_t)Nn * 256 * 4);
    size_t o_st  = take(512 * 4);
    size_t o_ss  = take(512 * 4);
    size_t o_pl  = take((size_t)G_ * 128 * 4);
    size_t o_ct  = take((size_t)G_ * 4);
    size_t o_xc  = take((size_t)G_ * 256 * 4);
    size_t o_t1  = take((size_t)G_ * 256 * 4);
    size_t o_t2  = take((size_t)G_ * 64 * 4);

    // per-call weight convert + transpose (f32 [K][N] -> bf16 [N][K])
    for (int b = 0; b < 2; b++) {
        int pb = 6 + b * 9;
        int nemb = FIN * 128;
        k_cvt_t<<<(nemb + 255) / 256, 256, 0, stream>>>(
            (const float*)d_in[pb + 0], (__bf16*)(ws + o_wbemb[b]), FIN, 128);
        for (int l = 0; l < L_; l++) {
            k_cvt_t<<<(128 * 256 + 255) / 256, 256, 0, stream>>>(
                (const float*)d_in[pb + 2] + (size_t)l * 128 * 256,
                (__bf16*)(ws + o_wbw1[b]) + (size_t)l * 256 * 128, 128, 256);
            k_cvt_t<<<(256 * 128 + 255) / 256, 256, 0, stream>>>(
                (const float*)d_in[pb + 6] + (size_t)l * 256 * 128,
                (__bf16*)(ws + o_wbw2[b]) + (size_t)l * 128 * 256, 256, 128);
        }
    }

    int nbg = (Nn + 127) / 128;
    float invN = 1.f / (float)Nn;
    float* h   = (float*)(ws + o_h);
    float* agg = (float*)(ws + o_agg);
    float* u   = (float*)(ws + o_u);
    float* st  = (float*)(ws + o_st);
    float* ssb = (float*)(ws + o_ss);

    for (int b = 0; b < 2; b++) {
        int pb = 6 + b * 9;
        const float* x    = (const float*)d_in[b * 3 + 0];
        const int* ei     = (const int*)d_in[b * 3 + 1];
        const int* batch  = (const int*)d_in[b * 3 + 2];

        k_emb<<<nbg, 256, 0, stream>>>(x, (const __bf16*)(ws + o_wbemb[b]),
                                       (const float*)d_in[pb + 1], h, Nn, FIN);
        for (int l = 0; l < L_; l++) {
            k_zero4<<<2048, 256, 0, stream>>>((float4*)agg, (long long)Nn * 32);
            k_zero4<<<1, 128, 0, stream>>>((float4*)st, 128);
            k_edge<<<2048, 256, 0, stream>>>(ei, h, agg, E_);
            k_mlp1<<<nbg, 256, 0, stream>>>(h, agg,
                (const __bf16*)(ws + o_wbw1[b]) + (size_t)l * 256 * 128,
                (const float*)d_in[pb + 3] + (size_t)l * 256,
                (const float*)d_in[pb + 8], l, u, st, Nn);
            k_bnfin<<<1, 256, 0, stream>>>(st,
                (const float*)d_in[pb + 4] + (size_t)l * 256,
                (const float*)d_in[pb + 5] + (size_t)l * 256, ssb, invN);
            k_mlp2<<<nbg, 256, 0, stream>>>(u, ssb,
                (const __bf16*)(ws + o_wbw2[b]) + (size_t)l * 128 * 256,
                (const float*)d_in[pb + 7] + (size_t)l * 128, h, Nn);
        }
        k_zero4<<<64, 256, 0, stream>>>((float4*)(ws + o_pl), (long long)G_ * 32);
        k_zero4<<<1, 64, 0, stream>>>((float4*)(ws + o_ct), (long long)G_ / 4);
        k_pool<<<2048, 256, 0, stream>>>(h, batch, (float*)(ws + o_pl), (float*)(ws + o_ct), Nn);
        k_poolfin<<<(G_ * 128 + 255) / 256, 256, 0, stream>>>((const float*)(ws + o_pl),
            (const float*)(ws + o_ct), (float*)(ws + o_xc), G_, b * 128);
    }
    k_head<<<1, 256, 0, stream>>>((const float*)(ws + o_xc),
        (const float*)d_in[24], (const float*)d_in[25],
        (const float*)d_in[26], (const float*)d_in[27],
        (const float*)d_in[28], (const float*)d_in[29],
        (float*)(ws + o_t1), (float*)(ws + o_t2), (float*)d_out, G_);
}
